// BboxIoULoss_2954937499989
// MI455X (gfx1250) — compile-verified, matrix operand order checked by reading, not run on hardware
//
#include <hip/hip_runtime.h>
#include <hip/hip_bf16.h>
#include <stdint.h>
#include <math.h>

#define ROWS 16
#define N_MAX 128
#define THR 0.5f

typedef __attribute__((ext_vector_type(16))) _Float16 v16h;
typedef __attribute__((ext_vector_type(8)))  float    v8f;
typedef __attribute__((ext_vector_type(4)))  unsigned v4u;
typedef __attribute__((ext_vector_type(8)))  int      v8i;
typedef __attribute__((ext_vector_type(4)))  int      v4i;

// ---------------------------------------------------------------------------
// TDM: async DMA of a rows x n f32 tile (row-major, stride n) global -> LDS.
// Descriptor layout per CDNA5 ISA ch.10.8 (D# groups 0/1; groups 2/3 zero
// since the tile is 2-D and we are not in a cluster).
// ---------------------------------------------------------------------------
__device__ __forceinline__ void tdm_load_tile_f32(const float* gptr,
                                                  unsigned lds_byte_addr,
                                                  unsigned n, unsigned rows) {
  uint64_t ga = (uint64_t)(uintptr_t)gptr;
  v4u g0;
  g0.x = 1u;                                            // count=1, user mode
  g0.y = lds_byte_addr;                                 // LDS destination
  g0.z = (unsigned)(ga & 0xFFFFFFFFu);                  // global_addr[31:0]
  g0.w = (unsigned)((ga >> 32) & 0x1FFFFFFu)            // global_addr[56:32]
       | (2u << 30);                                    // type = 2 ("image")

  v8i g1;
  g1[0] = (int)(2u << 16);                              // data_size=4B, wg_mask=0
  g1[1] = (int)((n & 0xFFFFu) << 16);                   // tensor_dim0 lo16
  g1[2] = (int)((n >> 16) | ((rows & 0xFFFFu) << 16));  // dim0 hi | dim1 lo
  g1[3] = (int)((rows >> 16) | ((n & 0xFFFFu) << 16));  // dim1 hi | tile_dim0
  g1[4] = (int)(rows & 0xFFFFu);                        // tile_dim1 (tile_dim2=0)
  g1[5] = (int)n;                                       // tensor_dim0_stride lo32
  g1[6] = 0;                                            // stride hi | dim1_stride lo
  g1[7] = 0;

  v4i gz = {0, 0, 0, 0};
#if defined(__clang_major__) && (__clang_major__ >= 23)
  v8i gz8 = {0, 0, 0, 0, 0, 0, 0, 0};
  __builtin_amdgcn_tensor_load_to_lds(g0, g1, gz, gz, gz8, 0);
#else
  __builtin_amdgcn_tensor_load_to_lds(g0, g1, gz, gz, 0);
#endif
}

// ---------------------------------------------------------------------------
// Sorted top-3 (value, payload) set, branchless: 3 v_cmp + 10 v_cndmask per
// insert, no EXEC-mask manipulation in the hot loop.
// ---------------------------------------------------------------------------
struct Top3 { float v0, v1, v2, d0, d1, d2; };

__device__ __forceinline__ void top3_insert(Top3& t, float v, float d) {
  bool  g0  = v > t.v0;
  float a0v = g0 ? v    : t.v0;  float a0d = g0 ? d    : t.d0;  // max(x, t0)
  float b0v = g0 ? t.v0 : v;     float b0d = g0 ? t.d0 : d;     // min(x, t0)
  bool  g1  = b0v > t.v1;
  float a1v = g1 ? b0v  : t.v1;  float a1d = g1 ? b0d  : t.d1;
  float b1v = g1 ? t.v1 : b0v;   float b1d = g1 ? t.d1 : b0d;
  bool  g2  = b1v > t.v2;
  float a2v = g2 ? b1v  : t.v2;  float a2d = g2 ? b1d  : t.d2;
  t.v0 = a0v; t.d0 = a0d;
  t.v1 = a1v; t.d1 = a1d;
  t.v2 = a2v; t.d2 = a2d;
}

__device__ __forceinline__ void top3_merge_xor(Top3& t, int mask) {
  float a0 = __shfl_xor(t.v0, mask), b0 = __shfl_xor(t.d0, mask);
  float a1 = __shfl_xor(t.v1, mask), b1 = __shfl_xor(t.d1, mask);
  float a2 = __shfl_xor(t.v2, mask), b2 = __shfl_xor(t.d2, mask);
  top3_insert(t, a0, b0);
  top3_insert(t, a1, b1);
  top3_insert(t, a2, b2);
}

// ---------------------------------------------------------------------------
// Main kernel: one 256-thread block per target row m.
// ---------------------------------------------------------------------------
__global__ void __launch_bounds__(256)
bbox_diou_loss_kernel(const float* __restrict__ out_moments,  // [S, P, 2]
                      const float* __restrict__ tgt_moments,  // [M, 2]
                      const int*   __restrict__ num_targets,  // [S]
                      const float* __restrict__ iou2ds,       // [M, N, N]
                      float* __restrict__ accum,              // [2]
                      int S, int P, int N) {
  __shared__ float tile[2][ROWS * N_MAX];
  __shared__ float waveRed[8][2];
  __shared__ float waveTop[8][6];
  __shared__ int   s_sample;

  const int m   = blockIdx.x;
  const int tid = threadIdx.x;

  if (tid == 0) {                       // repeat_interleave(arange(S), num_targets)[m]
    int s = 0, acc = 0;
    for (int k = 0; k < S; ++k) {
      int nt = num_targets[k];
      if (m >= acc && m < acc + nt) s = k;
      acc += nt;
    }
    s_sample = s;
  }

  const float* row = iou2ds + (size_t)m * N * N;
  const int nChunks = N / ROWS;

  // Prime the pipeline: DMA chunk 0 into buffer 0 (wave 0 only).
  if (tid < 32) {
    tdm_load_tile_f32(row, (unsigned)(uintptr_t)&tile[0][0], (unsigned)N, ROWS);
    __builtin_amdgcn_s_wait_tensorcnt(0);
  }
  __syncthreads();

  const int     s    = s_sample;
  const float2* om2  = (const float2*)(out_moments + (size_t)s * P * 2);
  const float   s2   = tgt_moments[2 * m];
  const float   e2   = tgt_moments[2 * m + 1];
  const float   len2 = e2 - s2;
  const float   ctr2 = s2 + e2;

  float lossAcc = 0.0f, cntAcc = 0.0f;
  Top3 t; t.v0 = t.v1 = t.v2 = -1.0f; t.d0 = t.d1 = t.d2 = 0.0f;

  for (int c = 0; c < nChunks; ++c) {
    // Overlap: DMA chunk c+1 into the other buffer while computing chunk c.
    if (c + 1 < nChunks && tid < 32) {
      tdm_load_tile_f32(row + (size_t)(c + 1) * ROWS * N,
                        (unsigned)(uintptr_t)&tile[(c + 1) & 1][0],
                        (unsigned)N, ROWS);
      // Prefetch the matching out_moments region toward L2/L0.
      int inext = (c + 1) * ROWS;
      __builtin_prefetch(&om2[inext * N - (inext * (inext - 1)) / 2], 0, 1);
    }

    const float* buf = &tile[c & 1][0];
    const int    i0  = c * ROWS;

    // Two rows in flight: threads 0-127 take even r, 128-255 take odd r.
    for (int r = (tid >> 7); r < ROWS; r += 2) {
      int i    = i0 + r;
      int len  = N - i;                                  // triu row length
      int base = i * N - (i * (i - 1)) / 2;              // flat p of (i, i)
      for (int t0 = (tid & 127); t0 < len; t0 += 128) {
        int    j    = i + t0;
        float  iou  = buf[r * N + j];
        float2 mo   = om2[base + t0];                    // one global_load_b64
        float  so   = mo.x, eo = mo.y;
        float inter = fmaxf(fminf(eo, e2) - fmaxf(so, s2), 0.0f);
        float uni   = (eo - so) + len2 - inter;
        float iou1  = inter / uni;
        float enc   = fmaxf(eo, e2) - fminf(so, s2);
        float cd    = 0.5f * ((so + eo) - ctr2);
        float diou  = iou1 - (cd * cd) / (enc * enc);
        float lossv = 1.0f - diou;
        bool  hit   = iou > THR;
        lossAcc += hit ? lossv : 0.0f;                   // cndmask + add
        cntAcc  += hit ? 1.0f  : 0.0f;
        top3_insert(t, iou, lossv);                      // branchless
      }
    }

    if (tid < 32 && c + 1 < nChunks) __builtin_amdgcn_s_wait_tensorcnt(0);
    __syncthreads();
  }

  // -------- wave reduction of (lossAcc, cntAcc): WMMA horizontal add --------
  // A = ones(16x32); B column n draws from lanes n and n+16, so with the lane
  // partial in f16 slot 0, D[*, n] = partial(n) + partial(n+16). One matrix op
  // replaces the first shuffle stage for each accumulator.
  v16h aOnes, bL, bC;
  v8f  zeroC;
#pragma unroll
  for (int k = 0; k < 16; ++k) { aOnes[k] = (_Float16)1.0f; bL[k] = (_Float16)0.0f; bC[k] = (_Float16)0.0f; }
#pragma unroll
  for (int k = 0; k < 8; ++k) zeroC[k] = 0.0f;
  bL[0] = (_Float16)lossAcc;
  bC[0] = (_Float16)cntAcc;

  v8f dL = __builtin_amdgcn_wmma_f32_16x16x32_f16(false, aOnes, false, bL,
                                                  (short)0, zeroC, false, false);
  v8f dC = __builtin_amdgcn_wmma_f32_16x16x32_f16(false, aOnes, false, bC,
                                                  (short)0, zeroC, false, false);
  float lsum = dL[0];   // partial(lane%16) + partial(lane%16 + 16)
  float csum = dC[0];
#pragma unroll
  for (int off = 8; off; off >>= 1) {
    lsum += __shfl_xor(lsum, off);
    csum += __shfl_xor(csum, off);
  }

  // -------- wave reduction of top-3: 5-step butterfly merge --------
#pragma unroll
  for (int mask = 16; mask; mask >>= 1) top3_merge_xor(t, mask);

  const int wave = tid >> 5, lane = tid & 31;
  if (lane == 0) {
    waveRed[wave][0] = lsum;  waveRed[wave][1] = csum;
    waveTop[wave][0] = t.v0;  waveTop[wave][1] = t.v1;  waveTop[wave][2] = t.v2;
    waveTop[wave][3] = t.d0;  waveTop[wave][4] = t.d1;  waveTop[wave][5] = t.d2;
  }
  __syncthreads();

  if (tid == 0) {
    float L = 0.0f, C = 0.0f;
    Top3 g; g.v0 = g.v1 = g.v2 = -1.0f; g.d0 = g.d1 = g.d2 = 0.0f;
    const int nWaves = blockDim.x >> 5;
    for (int w = 0; w < nWaves; ++w) {
      L += waveRed[w][0];
      C += waveRed[w][1];
      top3_insert(g, waveTop[w][0], waveTop[w][3]);
      top3_insert(g, waveTop[w][1], waveTop[w][4]);
      top3_insert(g, waveTop[w][2], waveTop[w][5]);
    }
    // Top-3 entries are always masked; add only those not already counted
    // by the iou > threshold pass.
    if (g.v0 <= THR) { L += g.d0; C += 1.0f; }
    if (g.v1 <= THR) { L += g.d1; C += 1.0f; }
    if (g.v2 <= THR) { L += g.d2; C += 1.0f; }
    atomicAdd(&accum[0], L);
    atomicAdd(&accum[1], C);
  }
}

__global__ void finalize_kernel(const float* __restrict__ accum,
                                float* __restrict__ out) {
  if (threadIdx.x == 0 && blockIdx.x == 0) out[0] = accum[0] / accum[1];
}

extern "C" void kernel_launch(void* const* d_in, const int* in_sizes, int n_in,
                              void* d_out, int out_size, void* d_ws, size_t ws_size,
                              hipStream_t stream) {
  const float* out_moments = (const float*)d_in[0];  // [S, P, 2]
  const float* tgt_moments = (const float*)d_in[1];  // [M, 2]
  const int*   num_targets = (const int*)d_in[2];    // [S]
  const float* iou2ds      = (const float*)d_in[3];  // [M, N, N]
  // d_in[4] (mask2d) is a static triu mask: the gather index is analytic.

  const int S  = in_sizes[2];
  const int M  = in_sizes[1] / 2;
  const int NN = in_sizes[3] / M;
  const int N  = (int)(sqrtf((float)NN) + 0.5f);
  const int P  = in_sizes[0] / (2 * S);

  float* accum = (float*)d_ws;                       // [loss_sum, mask_sum]
  hipMemsetAsync(accum, 0, 2 * sizeof(float), stream);

  bbox_diou_loss_kernel<<<M, 256, 0, stream>>>(out_moments, tgt_moments,
                                               num_targets, iou2ds, accum,
                                               S, P, N);
  finalize_kernel<<<1, 32, 0, stream>>>(accum, (float*)d_out);
}
